// DeepSeekV2MoeLayer_27805618275267
// MI455X (gfx1250) — matrix operand order repaired and checked by reading, not compiled, not run on hardware
//
#include <hip/hip_runtime.h>
#include <hip/hip_bf16.h>
#include <math.h>

typedef __attribute__((ext_vector_type(16))) _Float16 v16h;
typedef __attribute__((ext_vector_type(8)))  float    v8f;

#define T_TOK 2048
#define DH    2048
#define NE    16
#define II    1408
#define TOPK  6
#define MAXROWS 13312   // 6*T + 16*63 padding, rounded up

// workspace layout (bytes)
#define WS_COMBINE 0
#define WS_CNT     (WS_COMBINE + T_TOK*NE*4)   // 131072
#define WS_BASE    (WS_CNT + NE*4)
#define WS_TOK     (WS_BASE + NE*4)
#define WS_H       (WS_TOK + MAXROWS*4)

#define WMMA_F16(A, B, C) \
  __builtin_amdgcn_wmma_f32_16x16x32_f16(false, (A), false, (B), (short)0, (C), false, false)

__global__ __launch_bounds__(256) void k_zero(float* __restrict__ out, int n) {
  int i = blockIdx.x * 256 + threadIdx.x;
  if (i < n) out[i] = 0.f;
}

// one wave per token: fp32 dot over D for 16 experts, top-6, normalized weights
__global__ __launch_bounds__(256) void k_router(const float* __restrict__ x,
                                                const float* __restrict__ gw,
                                                float* __restrict__ combine) {
  int wid  = threadIdx.x >> 5;
  int lane = threadIdx.x & 31;
  int t = blockIdx.x * 8 + wid;
  const float* xr = x + (size_t)t * DH;
  float logit[NE];
  #pragma unroll
  for (int e = 0; e < NE; e++) {
    const float* gr = gw + (size_t)e * DH;
    float acc = 0.f;
    for (int d = lane; d < DH; d += 32) acc += xr[d] * gr[d];
    #pragma unroll
    for (int off = 16; off; off >>= 1) acc += __shfl_xor(acc, off, 32);
    logit[e] = acc;                 // broadcast to all lanes by butterfly
  }
  float m = logit[0];
  #pragma unroll
  for (int e = 1; e < NE; e++) m = fmaxf(m, logit[e]);
  float p[NE];
  #pragma unroll
  for (int e = 0; e < NE; e++) p[e] = __expf(logit[e] - m);
  // softmax denominator cancels in top-k renormalization
  unsigned mask = 0; float s = 0.f;
  #pragma unroll
  for (int k = 0; k < TOPK; k++) {
    int am = 0; float bv = -1.f;
    #pragma unroll
    for (int e = 0; e < NE; e++)
      if (!((mask >> e) & 1) && p[e] > bv) { bv = p[e]; am = e; }
    mask |= 1u << am; s += bv;
  }
  if (lane < NE) {
    float v = ((mask >> lane) & 1) ? p[lane] / s : 0.f;
    combine[(size_t)t * NE + lane] = v;
  }
}

__global__ __launch_bounds__(256) void k_count(const float* __restrict__ combine,
                                               int* __restrict__ cnt) {
  int e = blockIdx.x;
  __shared__ int c;
  if (threadIdx.x == 0) c = 0;
  __syncthreads();
  int local = 0;
  for (int t = threadIdx.x; t < T_TOK; t += 256)
    if (combine[(size_t)t * NE + e] > 0.f) local++;
  atomicAdd(&c, local);
  __syncthreads();
  if (threadIdx.x == 0) cnt[e] = c;
}

__global__ void k_scan(const int* __restrict__ cnt, int* __restrict__ base) {
  if (threadIdx.x == 0 && blockIdx.x == 0) {
    int r = 0;
    for (int e = 0; e < NE; e++) { base[e] = r; r += (cnt[e] + 63) & ~63; }
  }
}

// deterministic ordered compaction of token ids per expert + pad tile tail
__global__ __launch_bounds__(256) void k_compact(const float* __restrict__ combine,
                                                 const int* __restrict__ cnt,
                                                 const int* __restrict__ base,
                                                 int* __restrict__ tok) {
  int e = blockIdx.x;
  int b0 = base[e];
  __shared__ int sBase;
  __shared__ int wtot[8];
  if (threadIdx.x == 0) sBase = 0;
  __syncthreads();
  int wid = threadIdx.x >> 5, lane = threadIdx.x & 31;
  for (int t0 = 0; t0 < T_TOK; t0 += 256) {
    int t = t0 + threadIdx.x;
    bool pred = combine[(size_t)t * NE + e] > 0.f;
    unsigned long long bal = __ballot(pred);
    int rank = __popcll(bal & ((1ull << lane) - 1ull));
    if (lane == 0) wtot[wid] = __popcll(bal);
    __syncthreads();
    int wpre = 0, ctot = 0;
    #pragma unroll
    for (int w = 0; w < 8; w++) { if (w < wid) wpre += wtot[w]; ctot += wtot[w]; }
    if (pred) tok[b0 + sBase + wpre + rank] = t;
    __syncthreads();
    if (threadIdx.x == 0) sBase += ctot;
    __syncthreads();
  }
  int c = cnt[e];
  int pc = (c + 63) & ~63;
  for (int j = c + threadIdx.x; j < pc; j += 256) tok[b0 + j] = 0;
}

// up-projection + silu*mul: block = (expert, 64 gathered tokens, 128 i-cols)
// 8 waves as 2(M)x4(N); each wave: 32x32 for gate AND up.
// All 6 fragments (2 A + 4 B) are loaded into distinct registers, then the
// 8 WMMAs issue back-to-back: no WAR between a WMMA's B-read and the next
// fragment's cvt, so hazard NOPs are covered by real work.
__global__ __launch_bounds__(256) void k_up(const float* __restrict__ x,
                                            const float* __restrict__ w13,
                                            const int* __restrict__ cnt,
                                            const int* __restrict__ base,
                                            const int* __restrict__ tok,
                                            _Float16* __restrict__ hbuf) {
  int e = blockIdx.z;
  int c = cnt[e];
  int pc = (c + 63) & ~63;
  int mt = blockIdx.x;
  if (mt * 64 >= pc) return;                  // uniform per-block -> EXEC stays full
  int i0 = blockIdx.y * 128;
  int wid = threadIdx.x >> 5, lane = threadIdx.x & 31;
  int wm = wid & 1, wn = wid >> 1;
  int n0 = i0 + wn * 32;
  int b0 = base[e];
  int rb = mt * 64 + wm * 32;

  int tk0 = tok[b0 + rb + (lane & 15)];
  int tk1 = tok[b0 + rb + 16 + (lane & 15)];
  const float* xr0 = x + (size_t)tk0 * DH;
  const float* xr1 = x + (size_t)tk1 * DH;
  int kbA = (lane < 16) ? 0 : 16;
  int kbB = (lane < 16) ? 0 : 16;
  int nc0 = n0 + (lane & 15);
  const float* wg0 = w13 + ((size_t)e * 2 * II + nc0)      * DH + kbB;
  const float* wg1 = w13 + ((size_t)e * 2 * II + nc0 + 16) * DH + kbB;
  const float* wu0 = w13 + ((size_t)e * 2 * II + II + nc0)      * DH + kbB;
  const float* wu1 = w13 + ((size_t)e * 2 * II + II + nc0 + 16) * DH + kbB;

  v8f ag00 = {}, ag01 = {}, ag10 = {}, ag11 = {};
  v8f au00 = {}, au01 = {}, au10 = {}, au11 = {};
  for (int k0 = 0; k0 < DH; k0 += 32) {
    v16h a0, a1, bg0v, bg1v, bu0v, bu1v;
    #pragma unroll
    for (int j = 0; j < 8; j++) {
      a0[j]     = (_Float16)xr0[k0 + kbA + j];
      a0[8 + j] = (_Float16)xr0[k0 + kbA + 8 + j];
      a1[j]     = (_Float16)xr1[k0 + kbA + j];
      a1[8 + j] = (_Float16)xr1[k0 + kbA + 8 + j];
    }
    #pragma unroll
    for (int j = 0; j < 16; j++) {
      bg0v[j] = (_Float16)wg0[k0 + j];
      bg1v[j] = (_Float16)wg1[k0 + j];
      bu0v[j] = (_Float16)wu0[k0 + j];
      bu1v[j] = (_Float16)wu1[k0 + j];
    }
    ag00 = WMMA_F16(a0, bg0v, ag00);
    ag10 = WMMA_F16(a1, bg0v, ag10);
    ag01 = WMMA_F16(a0, bg1v, ag01);
    ag11 = WMMA_F16(a1, bg1v, ag11);
    au00 = WMMA_F16(a0, bu0v, au00);
    au10 = WMMA_F16(a1, bu0v, au10);
    au01 = WMMA_F16(a0, bu1v, au01);
    au11 = WMMA_F16(a1, bu1v, au11);
  }
  int rowbase = b0 + rb + ((lane < 16) ? 0 : 8);
  int colb = n0 + (lane & 15);
  #pragma unroll
  for (int v = 0; v < 8; v++) {
    size_t r0 = (size_t)(rowbase + v);            // rows [rb, rb+16)
    size_t r1 = (size_t)(rowbase + 16 + v);       // rows [rb+16, rb+32)
    float g, u, h;
    g = ag00[v]; u = au00[v];
    h = g * (1.f / (1.f + __expf(-g))) * u;
    hbuf[r0 * II + colb] = (_Float16)h;
    g = ag01[v]; u = au01[v];
    h = g * (1.f / (1.f + __expf(-g))) * u;
    hbuf[r0 * II + colb + 16] = (_Float16)h;
    g = ag10[v]; u = au10[v];
    h = g * (1.f / (1.f + __expf(-g))) * u;
    hbuf[r1 * II + colb] = (_Float16)h;
    g = ag11[v]; u = au11[v];
    h = g * (1.f / (1.f + __expf(-g))) * u;
    hbuf[r1 * II + colb + 16] = (_Float16)h;
  }
}

// down-projection: out[t,d] += combine[t,e] * (h_row . w2[e][d,:])
// wave computes 32 rows x 32 cols; loads 2 A + 2 B frags then 4 WMMAs back-to-back
__global__ __launch_bounds__(256) void k_down(const _Float16* __restrict__ hbuf,
                                              const float* __restrict__ w2,
                                              const float* __restrict__ combine,
                                              const int* __restrict__ cnt,
                                              const int* __restrict__ base,
                                              const int* __restrict__ tok,
                                              float* __restrict__ out) {
  int e = blockIdx.z;
  int c = cnt[e];
  int pc = (c + 63) & ~63;
  int mt = blockIdx.x;
  if (mt * 64 >= pc) return;
  int d0 = blockIdx.y * 128;
  int wid = threadIdx.x >> 5, lane = threadIdx.x & 31;
  int wm = wid & 1, wn = wid >> 1;
  int n0 = d0 + wn * 32;
  int b0 = base[e];
  int rb = mt * 64 + wm * 32;

  const _Float16* h0 = hbuf + (size_t)(b0 + rb + (lane & 15)) * II;
  const _Float16* h1 = hbuf + (size_t)(b0 + rb + 16 + (lane & 15)) * II;
  int kbA = (lane < 16) ? 0 : 16;
  int kbB = (lane < 16) ? 0 : 16;
  int nc0 = n0 + (lane & 15);
  const float* w20 = w2 + ((size_t)e * DH + nc0)      * II + kbB;
  const float* w21 = w2 + ((size_t)e * DH + nc0 + 16) * II + kbB;

  v8f a00 = {}, a01 = {}, a10 = {}, a11 = {};
  for (int k0 = 0; k0 < II; k0 += 32) {
    v16h af0, af1, b0v, b1v;
    #pragma unroll
    for (int j = 0; j < 8; j++) {
      af0[j]     = h0[k0 + kbA + j];
      af0[8 + j] = h0[k0 + kbA + 8 + j];
      af1[j]     = h1[k0 + kbA + j];
      af1[8 + j] = h1[k0 + kbA + 8 + j];
    }
    #pragma unroll
    for (int j = 0; j < 16; j++) {
      b0v[j] = (_Float16)w20[k0 + j];
      b1v[j] = (_Float16)w21[k0 + j];
    }
    a00 = WMMA_F16(af0, b0v, a00);
    a10 = WMMA_F16(af1, b0v, a10);
    a01 = WMMA_F16(af0, b1v, a01);
    a11 = WMMA_F16(af1, b1v, a11);
  }
  int lr0 = rb + ((lane < 16) ? 0 : 8);
  #pragma unroll
  for (int v = 0; v < 8; v++) {
    int lr = lr0 + v;
    if (lr < c) {
      int t = tok[b0 + lr];
      float w = combine[(size_t)t * NE + e];
      atomicAdd(&out[(size_t)t * DH + nc0],      w * a00[v]);
      atomicAdd(&out[(size_t)t * DH + nc0 + 16], w * a01[v]);
    }
    int lrh = lr + 16;
    if (lrh < c) {
      int t = tok[b0 + lrh];
      float w = combine[(size_t)t * NE + e];
      atomicAdd(&out[(size_t)t * DH + nc0],      w * a10[v]);
      atomicAdd(&out[(size_t)t * DH + nc0 + 16], w * a11[v]);
    }
  }
}

extern "C" void kernel_launch(void* const* d_in, const int* in_sizes, int n_in,
                              void* d_out, int out_size, void* d_ws, size_t ws_size,
                              hipStream_t stream) {
  (void)in_sizes; (void)n_in; (void)out_size; (void)ws_size;
  const float* x   = (const float*)d_in[0];
  const float* gw  = (const float*)d_in[1];
  const float* w13 = (const float*)d_in[2];
  const float* w2  = (const float*)d_in[3];
  // d_in[4] = top_k scalar; fixed to 6 at compile time
  float* out = (float*)d_out;
  char* ws = (char*)d_ws;
  float*    combine = (float*)(ws + WS_COMBINE);
  int*      cnt     = (int*)(ws + WS_CNT);
  int*      base    = (int*)(ws + WS_BASE);
  int*      tok     = (int*)(ws + WS_TOK);
  _Float16* hbuf    = (_Float16*)(ws + WS_H);

  k_zero<<<(T_TOK * DH + 255) / 256, 256, 0, stream>>>(out, T_TOK * DH);
  k_router<<<T_TOK / 8, 256, 0, stream>>>(x, gw, combine);
  k_count<<<NE, 256, 0, stream>>>(combine, cnt);
  k_scan<<<1, 1, 0, stream>>>(cnt, base);
  k_compact<<<NE, 256, 0, stream>>>(combine, cnt, base, tok);
  dim3 gu(32, II / 128, NE);   // m-tile fastest -> weight stripes stay L2-hot
  k_up<<<gu, 256, 0, stream>>>(x, w13, cnt, base, tok, hbuf);
  dim3 gd(32, DH / 128, NE);
  k_down<<<gd, 256, 0, stream>>>(hbuf, w2, combine, cnt, base, tok, out);
}
